// MaskYolo_14070312861962
// MI455X (gfx1250) — compile-verified, matrix operand order checked by reading, not run on hardware
//
#include <hip/hip_runtime.h>

// ---------------------------------------------------------------------------
// MaskYolo pipeline for MI455X (gfx1250, wave32, WMMA).
// Heavy stage = implicit GEMM conv 256->128 (3x3) via v_wmma_f32_16x16x32_bf16,
// register-blocked: 7 M-tiles share 8 B fragments held in 64 VGPRs per tap.
// 1x1 conv folded into the WMMA epilogue via shfl_xor + ds_add_f32.
// ---------------------------------------------------------------------------

typedef __bf16 bf16_t;
typedef float  v8f   __attribute__((ext_vector_type(8)));
typedef bf16_t v8bf  __attribute__((ext_vector_type(8)));
typedef bf16_t v16bf __attribute__((ext_vector_type(16)));

#define NANCH   340          // 256+64+16+4 anchors per batch
#define TOPK_   64
#define MASKSZ  28
#define NPIX    784          // 28*28
#define CIN     256
#define COUT    128
#define PATCHSZ 512
#define NSAMP   128          // B * TOPK
#define PADW    30           // 28 + 1px zero halo each side
#define PATCH_ELEMS (PADW*PADW*CIN)   // 230400 bf16 per sample
#define MTC     7            // M-tiles per register-blocked chunk (49 = 7*7)

// ws layout (bytes)
#define WS_BOXES  0          // 2*64*4 f32
#define WS_BOXI   2048       // 2*64*4 i32 (clipped int boxes)
#define WS_KEEP   4096       // 128 i32
#define WS_BW     8192       // packed bf16 weights: 9*8*8*32*16 = 294912 elems
#define WS_PATCH  (1u<<20)   // 128 * 230400 * 2 B = ~59 MB of bf16 patches

// d_out layout (elements): scores[128] boxes[512] labels[128] masks[2*64*512*512] keep[128]
#define OUT_BOXES 128
#define OUT_LAB   640
#define OUT_MASK  768
#define OUT_KEEP  33555200

__device__ __forceinline__ float sigm(float x) { return 1.f / (1.f + __expf(-x)); }

// ---------------------------------------------------------------------------
// Kernel 1: decode + top-64 + NMS.  One block per batch image.  Tiny.
// ---------------------------------------------------------------------------
__global__ __launch_bounds__(256)
void k_decode_nms(const float* __restrict__ p0, const float* __restrict__ p1,
                  const float* __restrict__ p2, const float* __restrict__ p3,
                  float* __restrict__ dout, unsigned char* __restrict__ ws)
{
    const int b = blockIdx.x;
    const int t = threadIdx.x;

    __shared__ float sc[NANCH];
    __shared__ float bxs[NANCH][4];
    __shared__ float rv[256];
    __shared__ int   ri[256];
    __shared__ int   sel_idx[TOPK_];
    __shared__ float sel_val[TOPK_];
    __shared__ float sel_box[TOPK_][4];
    __shared__ int   keepf[TOPK_];

    // decode all anchors
    for (int a = t; a < NANCH; a += 256) {
        const float* p; int H, pos, stride;
        if      (a < 256) { p = p0; H = 16; pos = a;       stride = 32;  }
        else if (a < 320) { p = p1; H = 8;  pos = a - 256; stride = 64;  }
        else if (a < 336) { p = p2; H = 4;  pos = a - 320; stride = 128; }
        else              { p = p3; H = 2;  pos = a - 336; stride = 256; }
        const int HW = H * H;
        const float* pb = p + (size_t)b * 6 * HW;
        float tx = pb[0*HW+pos], ty = pb[1*HW+pos];
        float tw = pb[2*HW+pos], th = pb[3*HW+pos];
        float obj = pb[4*HW+pos];
        int gy = pos / H, gx = pos % H;
        float cx = (tx + gx) * (float)stride, cy = (ty + gy) * (float)stride;
        float w  = __expf(tw) * (float)stride, h = __expf(th) * (float)stride;
        bxs[a][0] = cx - 0.5f*w;  bxs[a][1] = cy - 0.5f*h;
        bxs[a][2] = cx + 0.5f*w;  bxs[a][3] = cy + 0.5f*h;
        float s = sigm(obj);
        sc[a] = (s > 0.5f) ? s : 0.0f;     // score mask (SCORE_TH)
    }
    __syncthreads();

    // iterative argmax top-64 (stable: ties -> lower index, matching lax.top_k)
    for (int round = 0; round < TOPK_; ++round) {
        float bv = -1.f; int bi = 0;
        for (int a = t; a < NANCH; a += 256) {
            float v = sc[a];
            if (v > bv) { bv = v; bi = a; }
        }
        rv[t] = bv; ri[t] = bi;
        __syncthreads();
        for (int off = 128; off > 0; off >>= 1) {
            if (t < off) {
                float v2 = rv[t+off]; int i2 = ri[t+off];
                if (v2 > rv[t] || (v2 == rv[t] && i2 < ri[t])) { rv[t] = v2; ri[t] = i2; }
            }
            __syncthreads();
        }
        if (t == 0) {
            sel_idx[round] = ri[0];
            sel_val[round] = fmaxf(rv[0], 0.f);
            sc[ri[0]] = -1.f;
        }
        __syncthreads();
    }

    if (t < TOPK_) {
        int a = sel_idx[t];
        sel_box[t][0] = bxs[a][0]; sel_box[t][1] = bxs[a][1];
        sel_box[t][2] = bxs[a][2]; sel_box[t][3] = bxs[a][3];
        keepf[t] = (sel_val[t] > 0.5f) ? 1 : 0;
    }
    __syncthreads();

    // sequential NMS (single class: pred has exactly one class channel -> label 0)
    if (t == 0) {
        for (int i = 0; i < TOPK_; ++i) {
            if (!keepf[i]) continue;
            float ax0=sel_box[i][0], ay0=sel_box[i][1], ax1=sel_box[i][2], ay1=sel_box[i][3];
            float aarea = fmaxf(ax1-ax0,0.f)*fmaxf(ay1-ay0,0.f);
            for (int j = i+1; j < TOPK_; ++j) {
                if (!keepf[j]) continue;
                float qx0=sel_box[j][0], qy0=sel_box[j][1], qx1=sel_box[j][2], qy1=sel_box[j][3];
                float barea = fmaxf(qx1-qx0,0.f)*fmaxf(qy1-qy0,0.f);
                float ix0=fmaxf(ax0,qx0), iy0=fmaxf(ay0,qy0);
                float ix1=fminf(ax1,qx1), iy1=fminf(ay1,qy1);
                float inter = fmaxf(ix1-ix0,0.f)*fmaxf(iy1-iy0,0.f);
                float iou = inter / fmaxf(aarea + barea - inter, 1e-9f);
                if (iou > 0.5f) keepf[j] = 0;
            }
        }
    }
    __syncthreads();

    float* wsBox  = (float*)(ws + WS_BOXES);
    int*   wsBoxI = (int*)  (ws + WS_BOXI);
    int*   wsKeep = (int*)  (ws + WS_KEEP);
    if (t < TOPK_) {
        int idx = b*TOPK_ + t;
        int k = keepf[t];
        dout[idx] = sel_val[t] * (float)k;                   // out_scores
        for (int c2 = 0; c2 < 4; ++c2) {
            float v = sel_box[t][c2];
            dout[OUT_BOXES + idx*4 + c2] = v;                // boxes_k
            wsBox[idx*4 + c2] = v;
            float bc = fminf(fmaxf(v, 0.f), (float)(PATCHSZ-1));
            wsBoxI[idx*4 + c2] = (int)bc;                    // clipped int box
        }
        ((int*)dout)[OUT_LAB + idx] = 0;                     // labels (1 class)
        dout[OUT_KEEP + idx] = (float)k;                     // keep
        wsKeep[idx] = k;
    }
}

// ---------------------------------------------------------------------------
// Kernel 2: pack w1 (O=128, I=256, 3, 3) f32 -> bf16 WMMA B fragments.
// Fragment f = (tap, kslice, ntile); lane l holds column N = nt*16 + (l&15),
// values j0..7  -> K = base+{0..7},  j8..15 -> K = base+{16..23},
// with base = (l>=16 ? 8 : 0), mirroring the 16-bit A layout (ISA 7.12.2).
// ---------------------------------------------------------------------------
__global__ __launch_bounds__(256)
void k_pack_w(const float* __restrict__ w1, bf16_t* __restrict__ Bw)
{
    int idx = blockIdx.x*256 + threadIdx.x;
    if (idx >= 9*8*8*32*16) return;
    int j   =  idx        & 15;
    int l   = (idx >> 4)  & 31;
    int nt  = (idx >> 9)  & 7;
    int ks  = (idx >> 12) & 7;
    int tap =  idx >> 15;                       // 0..8 = ky*3+kx
    int klocal = (j < 8 ? j : j + 8) + ((l >= 16) ? 8 : 0);
    int c = ks*32 + klocal;                     // input channel
    int n = nt*16 + (l & 15);                   // output channel
    float v = w1[(size_t)n*2304 + (size_t)c*9 + tap];
    Bw[idx] = (bf16_t)v;
}

// ---------------------------------------------------------------------------
// Kernel 3: ROI-align one box per block into a zero-haloed 30x30x256 bf16
// patch, channel-contiguous so GEMM A fragments are contiguous 16B loads.
// Nearest-upsampling of feat1/2/3 folded into index shifts (>>1,>>2,>>3).
// ---------------------------------------------------------------------------
__global__ __launch_bounds__(256)
void k_roi(const float* __restrict__ f0, const float* __restrict__ f1,
           const float* __restrict__ f2, const float* __restrict__ f3,
           const float* __restrict__ wsBox, bf16_t* __restrict__ patches)
{
    const int s = blockIdx.x;                 // 0..127  (b = s>>6, roi = s&63)
    const int b = s >> 6;
    const int t = threadIdx.x;
    bf16_t* P = patches + (size_t)s * PATCH_ELEMS;

    // zero whole padded patch (halo included)
    unsigned int* Pz = (unsigned int*)P;
    for (int i = t; i < PATCH_ELEMS/2; i += 256) Pz[i] = 0u;

    __shared__ int   sx0[NPIX], sy0[NPIX];
    __shared__ float sfx[NPIX], sfy[NPIX];
    const float* box = wsBox + s*4;
    float b0 = box[0]*0.25f, b1v = box[1]*0.25f, b2v = box[2]*0.25f, b3v = box[3]*0.25f;
    float bw = (b2v - b0)  * (1.0f/28.0f);
    float bh = (b3v - b1v) * (1.0f/28.0f);
    for (int p = t; p < NPIX; p += 256) {
        int py = p / 28, px = p % 28;
        float x = fminf(fmaxf(b0  + (px + 0.5f)*bw, 0.f), 127.f);
        float y = fminf(fmaxf(b1v + (py + 0.5f)*bh, 0.f), 127.f);
        int x0 = (int)floorf(x), y0 = (int)floorf(y);
        sx0[p] = x0; sy0[p] = y0;
        sfx[p] = x - (float)x0; sfy[p] = y - (float)y0;
    }
    __syncthreads();

    for (int i = t; i < NPIX*CIN; i += 256) {
        int c = i & 255;
        int p = i >> 8;
        int x0 = sx0[p], y0 = sy0[p];
        int x1 = min(x0+1, 127), y1 = min(y0+1, 127);
        float fx = sfx[p], fy = sfy[p];
        const float* F; int dim, sh;
        if      (c < 64)  { F = f0 + ((size_t)b*64 +  c      )*128*128; sh = 0; dim = 128; }
        else if (c < 128) { F = f1 + ((size_t)b*64 + (c-64) )*64*64;    sh = 1; dim = 64;  }
        else if (c < 192) { F = f2 + ((size_t)b*64 + (c-128))*32*32;    sh = 2; dim = 32;  }
        else              { F = f3 + ((size_t)b*64 + (c-192))*16*16;    sh = 3; dim = 16;  }
        int X0 = x0 >> sh, X1 = x1 >> sh, Y0 = y0 >> sh, Y1 = y1 >> sh;
        float v00 = F[Y0*dim + X0], v01 = F[Y0*dim + X1];
        float v10 = F[Y1*dim + X0], v11 = F[Y1*dim + X1];
        float v = v00*(1.f-fy)*(1.f-fx) + v01*(1.f-fy)*fx
                + v10*fy*(1.f-fx)       + v11*fy*fx;
        int py = p / 28, px = p % 28;
        P[((size_t)(py+1)*PADW + (px+1))*CIN + c] = (bf16_t)v;
    }
}

// ---------------------------------------------------------------------------
// Kernel 4: implicit-GEMM conv (9 shifted GEMMs, K=2304 total) with
// v_wmma_f32_16x16x32_bf16, register-blocked over M:
//   chunk = 7 M-tiles; per tap, the 8 B fragments are loaded once into
//   64 VGPRs and reused by 7x8 = 56 WMMAs.  Tap/chunk loops are kept as
//   real loops (unroll disabled) so live state = 56 acc + 64 B + A staging
//   VGPRs -> no spills, and A loads pipeline against WMMAs.
// Epilogue fuses bias + SiLU + 1x1 conv: a pixel's 16 channels live in one
// 16-lane half of the D fragment -> 4x shfl_xor butterfly + ds_add_f32 into
// an LDS logit array.  Then sigmoid + 512x512 binary paste.  ~7 KB LDS.
// ---------------------------------------------------------------------------
__global__ __launch_bounds__(256)
void k_conv_paste(const bf16_t* __restrict__ patches, const bf16_t* __restrict__ Bw,
                  const float* __restrict__ b1, const float* __restrict__ w2,
                  const float* __restrict__ b2,
                  const int* __restrict__ wsBoxI, const int* __restrict__ wsKeep,
                  float* __restrict__ dout)
{
    __shared__ float logit[NPIX];
    __shared__ float soft[NPIX];
    __shared__ float w2s[COUT];

    const int s    = blockIdx.x;
    const int t    = threadIdx.x;
    const int lane = t & 31;
    const int wave = t >> 5;           // N tile 0..7
    const int nrow = lane & 15;        // M row (A) / N col (B,C,D) within tile
    const int hi   = lane >> 4;        // K half selector

    if (t < COUT) w2s[t] = w2[t];
    const float bb2 = b2[0];
    for (int p = t; p < NPIX; p += 256) logit[p] = bb2;
    __syncthreads();

    const bf16_t* P = patches + (size_t)s * PATCH_ELEMS;
    const int   n    = wave*16 + nrow;
    const float bias = b1[n];
    const float wn   = w2s[n];

    #pragma clang loop unroll(disable)
    for (int mtc = 0; mtc < 7; ++mtc) {            // 7 chunks of 7 M-tiles
        v8f acc[MTC];
        int rowoff[MTC];
        #pragma unroll
        for (int u = 0; u < MTC; ++u) {
            acc[u] = (v8f){};
            int p = (mtc*MTC + u)*16 + nrow;       // this lane's A-row pixel
            rowoff[u] = (p / 28)*PADW + (p % 28);  // padded-patch row base
        }
        #pragma clang loop unroll(disable)
        for (int tap = 0; tap < 9; ++tap) {        // 3x3 shifted GEMMs
            const int tshift = (tap/3)*PADW + (tap%3);
            // load this tap's 8 B fragments into registers (reused by 7 tiles)
            v16bf bfrag[8];
            #pragma unroll
            for (int ks = 0; ks < 8; ++ks) {
                const bf16_t* Bp = Bw + ((((size_t)tap*8 + ks)*8 + wave)*32 + lane)*16;
                v8bf q0 = *(const v8bf*)(Bp);
                v8bf q1 = *(const v8bf*)(Bp + 8);
                bfrag[ks] = __builtin_shufflevector(q0, q1,
                    0,1,2,3,4,5,6,7,8,9,10,11,12,13,14,15);
            }
            #pragma unroll
            for (int u = 0; u < MTC; ++u) {
                const bf16_t* Abase = P + (size_t)(rowoff[u] + tshift)*CIN + hi*8;
                #pragma unroll
                for (int ks = 0; ks < 8; ++ks) {   // K slices of 32 channels
                    v8bf a0 = *(const v8bf*)(Abase + ks*32);
                    v8bf a1 = *(const v8bf*)(Abase + ks*32 + 16);
                    v16bf av = __builtin_shufflevector(a0, a1,
                        0,1,2,3,4,5,6,7,8,9,10,11,12,13,14,15);
                    acc[u] = __builtin_amdgcn_wmma_f32_16x16x32_bf16(
                            false, av, false, bfrag[ks], (short)0, acc[u], false, false);
                }
            }
        }
        // Epilogue: D layout -> vgpr r, lanes0-15: (M=tile*16+r, N=n),
        // lanes16-31: (M=tile*16+r+8, N=n).  Fuse bias+SiLU+w2 dot; reduce
        // each pixel's 16 channels across its 16-lane half, accumulate LDS.
        #pragma unroll
        for (int u = 0; u < MTC; ++u) {
            #pragma unroll
            for (int r = 0; r < 8; ++r) {
                float h = acc[u][r] + bias;
                float v = h * sigm(h) * wn;          // silu(h) * w2[n]
                v += __shfl_xor(v, 8, 32);
                v += __shfl_xor(v, 4, 32);
                v += __shfl_xor(v, 2, 32);
                v += __shfl_xor(v, 1, 32);
                if (nrow == 0)
                    atomicAdd(&logit[(mtc*MTC + u)*16 + r + hi*8], v); // ds_add_f32
            }
        }
    }
    __syncthreads();

    for (int pix = t; pix < NPIX; pix += 256)
        soft[pix] = sigm(logit[pix]);
    __syncthreads();

    // paste 28x28 soft mask -> 512x512 binary mask (dominant HBM write: 134MB)
    const int bx0 = wsBoxI[s*4+0], by0 = wsBoxI[s*4+1];
    const int bx1 = wsBoxI[s*4+2], by1 = wsBoxI[s*4+3];
    const int bwid = bx1 - bx0 + 1, bhei = by1 - by0 + 1;
    const int kp = wsKeep[s];
    float* om = dout + OUT_MASK + (size_t)s * (PATCHSZ*PATCHSZ);
    for (int i = t; i < PATCHSZ*PATCHSZ; i += 256) {
        int Y = i >> 9, X = i & 511;
        float v = 0.f;
        if (kp && Y >= by0 && Y <= by1 && X >= bx0 && X <= bx1) {
            int iy = min(max((Y - by0) * MASKSZ / bhei, 0), MASKSZ-1);
            int ix = min(max((X - bx0) * MASKSZ / bwid, 0), MASKSZ-1);
            v = (soft[iy*MASKSZ + ix] > 0.5f) ? 1.f : 0.f;
        }
        om[i] = v;
    }
}

// ---------------------------------------------------------------------------
extern "C" void kernel_launch(void* const* d_in, const int* in_sizes, int n_in,
                              void* d_out, int out_size, void* d_ws, size_t ws_size,
                              hipStream_t stream)
{
    const float* p0 = (const float*)d_in[0];
    const float* p1 = (const float*)d_in[1];
    const float* p2 = (const float*)d_in[2];
    const float* p3 = (const float*)d_in[3];
    const float* f0 = (const float*)d_in[4];
    const float* f1 = (const float*)d_in[5];
    const float* f2 = (const float*)d_in[6];
    const float* f3 = (const float*)d_in[7];
    const float* w1 = (const float*)d_in[8];
    const float* b1 = (const float*)d_in[9];
    const float* w2 = (const float*)d_in[10];
    const float* b2 = (const float*)d_in[11];

    unsigned char* ws = (unsigned char*)d_ws;
    float* out = (float*)d_out;

    // 1) decode + topk + NMS (also writes scores/boxes/labels/keep outputs)
    k_decode_nms<<<2, 256, 0, stream>>>(p0, p1, p2, p3, out, ws);

    // 2) pack conv weights into bf16 WMMA B-fragment layout (294912 elems)
    k_pack_w<<<(9*8*8*32*16 + 255)/256, 256, 0, stream>>>(w1, (bf16_t*)(ws + WS_BW));

    // 3) ROI-align each of the 128 boxes into a padded bf16 patch
    k_roi<<<NSAMP, 256, 0, stream>>>(f0, f1, f2, f3,
                                     (const float*)(ws + WS_BOXES),
                                     (bf16_t*)(ws + WS_PATCH));

    // 4) WMMA conv + fused SiLU/1x1-conv epilogue + sigmoid + paste
    k_conv_paste<<<NSAMP, 256, 0, stream>>>(
        (const bf16_t*)(ws + WS_PATCH), (const bf16_t*)(ws + WS_BW),
        b1, w2, b2,
        (const int*)(ws + WS_BOXI), (const int*)(ws + WS_KEEP), out);
}
